// WindFarmGNN_71725953843308
// MI455X (gfx1250) — matrix operand order repaired and checked by reading, not compiled, not run on hardware
//
#include <hip/hip_runtime.h>
#include <hip/hip_bf16.h>
#include <math.h>

// ---------------------------------------------------------------------------
// WindFarmGNN fully fused: kNN + edge features + 3x GATv2 layer, one
// workgroup per batch element, all intermediates in LDS, dense node
// transforms on v_wmma_f32_16x16x32_f16.
// ---------------------------------------------------------------------------

typedef __attribute__((ext_vector_type(16))) _Float16 v16h;
typedef __attribute__((ext_vector_type(8)))  _Float16 v8h;
typedef __attribute__((ext_vector_type(8)))  float    v8f;

#define GNN_B     1024
#define GNN_N     100
#define GNN_K     5
#define GNN_H     64
#define GNN_MPAD  112      // 7 tiles of 16 rows
#define XS16      72       // f16 activation row stride (bank skew; 144B, 16B-mult)
#define WS16      72       // f16 transposed-weight row stride
#define XS32      68       // f32 activation row stride
#define NTHREADS  256

struct GParams {
    const float* We[3];
    const float* Wl[3];
    const float* Wr[3];
    const float* att[3];
    const float* bl[3];
    const float* bo[3];
    const float* br[3];
};

// ---- WMMA fragment helpers (ISA 7.12.2 layouts, wave32) --------------------

// A: 16x32 f16 (MxK). lanes 0-15: M=lane, K={0-7,16-23}; lanes 16-31: M=lane-16,
// K={8-15,24-31}. Two contiguous 16B chunks per lane.
__device__ __forceinline__ v16h frag_A(const _Float16* xbase, int mtile, int ktile, int lane) {
    const int l = lane & 15, grp = lane >> 4;
    const _Float16* p = xbase + (mtile * 16 + l) * XS16 + ktile * 32 + grp * 8;
    v8h lo = *(const v8h*)(p);
    v8h hi = *(const v8h*)(p + 16);
    v16h a;
#pragma unroll
    for (int i = 0; i < 8; ++i) { a[i] = lo[i]; a[i + 8] = hi[i]; }
    return a;
}

// B: 32x16 f16 (KxN). lane: N = lane&15; lanes 0-15: K={0-7,16-23}; lanes 16-31: K={8-15,24-31}.
// Weights stored N-major (transposed) in LDS -> two contiguous 8-f16 chunks per lane.
__device__ __forceinline__ v16h frag_B(const _Float16* wt, int ntile, int ktile, int lane) {
    const int l = lane & 15, grp = lane >> 4;
    const _Float16* p = wt + (ntile * 16 + l) * WS16 + ktile * 32 + grp * 8;
    v8h lo = *(const v8h*)(p);
    v8h hi = *(const v8h*)(p + 16);
    v16h b;
#pragma unroll
    for (int i = 0; i < 8; ++i) { b[i] = lo[i]; b[i + 8] = hi[i]; }
    return b;
}

// C/D: 16x16 f32, 8 VGPRs. VGPR r: M = r + (lane>=16)*8, N = lane&15.
__device__ __forceinline__ void store_C(float* dst, v8f c, int mtile, int ntile,
                                        int lane, const float* bias) {
    const int l = lane & 15, grp = lane >> 4;
    const float bv = bias[ntile * 16 + l];
#pragma unroll
    for (int r = 0; r < 8; ++r) {
        dst[(mtile * 16 + grp * 8 + r) * XS32 + ntile * 16 + l] = c[r] + bv;
    }
}

// ---------------------------------------------------------------------------

__global__ void __launch_bounds__(NTHREADS)
windfarm_gnn_fused(const float* __restrict__ wdir,
                   const float* __restrict__ wspd,
                   const float* __restrict__ yawp,
                   const float* __restrict__ layout,
                   GParams P,
                   float* __restrict__ out) {
    __shared__ float s_pos[2 * GNN_N];
    __shared__ float s_wind[2 * GNN_N];
    __shared__ int   s_nbr[GNN_N * GNN_K];
    __shared__ float s_edge[GNN_N * GNN_K * 5];
    __shared__ __align__(16) _Float16 s_x16[GNN_MPAD * XS16];   // layer input (f16)
    __shared__ __align__(16) _Float16 s_WtL[GNN_H * WS16];      // Wl^T, K-padded
    __shared__ __align__(16) _Float16 s_WtR[GNN_H * WS16];      // Wr^T, K-padded
    __shared__ float s_We[5 * GNN_H];
    __shared__ float s_att[GNN_H], s_bl[GNN_H], s_br[GNN_H], s_bo[GNN_H];
    __shared__ float s_xl[GNN_MPAD * XS32];                     // x @ Wl + bl (f32)
    __shared__ float s_xr[GNN_MPAD * XS32];                     // x @ Wr + br (f32)
    __shared__ float s_alpha[GNN_N * GNN_K];                    // logits -> alpha

    const int tid  = threadIdx.x;
    const int lane = tid & 31;
    const int wave = tid >> 5;
    const int b    = blockIdx.x;

    // ---- Phase 0: node features ------------------------------------------
    float ws_n = 0.0f, yw_r = 0.0f;
    if (tid < GNN_N) {
        const int g = b * GNN_N + tid;
        const float d2r = 0.017453292519943295f;
        float wd = wdir[g] * d2r;
        ws_n = wspd[g] * (1.0f / 28.0f);
        yw_r = yawp[g] * d2r;
        float px = layout[2 * g + 0] * (2.0f / 5000.0f) - 1.0f;
        float py = layout[2 * g + 1] * (2.0f / 5000.0f) - 1.0f;
        s_pos[2 * tid + 0] = px;
        s_pos[2 * tid + 1] = py;
        s_wind[2 * tid + 0] = ws_n * __cosf(wd);
        s_wind[2 * tid + 1] = ws_n * __sinf(wd);
    }
    // zero the padded f16 activation tile (rows 100..111 and cols >= din stay 0)
    for (int idx = tid; idx < GNN_MPAD * XS16; idx += NTHREADS)
        s_x16[idx] = (_Float16)0.0f;
    __syncthreads();
    if (tid < GNN_N) {
        s_x16[tid * XS16 + 0] = (_Float16)ws_n;
        s_x16[tid * XS16 + 1] = (_Float16)yw_r;
    }
    __syncthreads();

    // ---- Phase 1: brute-force kNN (incl. self) + edge features -----------
    if (tid < GNN_N) {
        const float px = s_pos[2 * tid + 0], py = s_pos[2 * tid + 1];
        float bd[GNN_K];
        int   bi[GNN_K];
#pragma unroll
        for (int k = 0; k < GNN_K; ++k) { bd[k] = 3.0e38f; bi[k] = 0; }
        for (int j = 0; j < GNN_N; ++j) {
            float dx = px - s_pos[2 * j + 0];
            float dy = py - s_pos[2 * j + 1];
            float d2 = dx * dx + dy * dy;
            if (d2 < bd[GNN_K - 1]) {
                int p = GNN_K - 1;
                while (p > 0 && d2 < bd[p - 1]) {
                    bd[p] = bd[p - 1]; bi[p] = bi[p - 1]; --p;
                }
                bd[p] = d2; bi[p] = j;
            }
        }
        const float wdx = s_wind[2 * tid + 0], wdy = s_wind[2 * tid + 1];
#pragma unroll
        for (int k = 0; k < GNN_K; ++k) {
            const int j = bi[k];
            s_nbr[tid * GNN_K + k] = j;
            float pdx = px - s_pos[2 * j + 0];   // pos[dst] - pos[src]
            float pdy = py - s_pos[2 * j + 1];
            float wsx = s_wind[2 * j + 0], wsy = s_wind[2 * j + 1];
            float* e = &s_edge[(tid * GNN_K + k) * 5];
            e[0] = __fsqrt_rn(pdx * pdx + pdy * pdy);
            e[1] = wsx * pdx + wsy * pdy;        // dot_s
            e[2] = wsx * pdy - wsy * pdx;        // cross_s
            e[3] = wdx * pdx + wdy * pdy;        // dot_d
            e[4] = wdx * pdy - wdy * pdx;        // cross_d
        }
    }
    __syncthreads();

    // ---- GATv2 layers -----------------------------------------------------
#pragma unroll 1
    for (int l = 0; l < 3; ++l) {
        const int din    = (l == 0) ? 2 : GNN_H;
        const int kTiles = (l == 0) ? 1 : 2;

        // load per-layer params: W^T (f16, zero-padded in K), We/att/biases (f32)
        const float* Wlg = P.Wl[l];
        const float* Wrg = P.Wr[l];
        for (int idx = tid; idx < GNN_H * GNN_H; idx += NTHREADS) {
            const int n = idx >> 6, k = idx & 63;
            const float vl = (k < din) ? Wlg[k * GNN_H + n] : 0.0f;
            const float vr = (k < din) ? Wrg[k * GNN_H + n] : 0.0f;
            s_WtL[n * WS16 + k] = (_Float16)vl;
            s_WtR[n * WS16 + k] = (_Float16)vr;
        }
        for (int idx = tid; idx < 5 * GNN_H; idx += NTHREADS) s_We[idx] = P.We[l][idx];
        if (tid < GNN_H) {
            s_att[tid] = P.att[l][tid];
            s_bl[tid]  = P.bl[l][tid];
            s_br[tid]  = P.br[l][tid];
            s_bo[tid]  = P.bo[l][tid];
        }
        __syncthreads();

        // Dense transforms on the matrix pipe: 7 Mtiles x 4 Ntiles x {Wl,Wr}
        // = 56 tiles over 8 waves, wave-uniform loop (EXEC all-ones for WMMA).
        for (int t = wave * 7; t < wave * 7 + 7; ++t) {
            const int sel   = t / 28;              // 0 -> xl, 1 -> xr
            const int rem   = t % 28;
            const int mtile = rem >> 2;
            const int ntile = rem & 3;
            const _Float16* wt  = sel ? s_WtR : s_WtL;
            const float*   bias = sel ? s_br  : s_bl;
            float*         dst  = sel ? s_xr  : s_xl;
            v8f c = {};
            for (int kt = 0; kt < kTiles; ++kt) {
                v16h a = frag_A(s_x16, mtile, kt, lane);
                v16h bm = frag_B(wt, ntile, kt, lane);
                c = __builtin_amdgcn_wmma_f32_16x16x32_f16(
                        /*neg_a=*/false, a, /*neg_b=*/false, bm,
                        /*c_mod=*/(short)0, c, /*reuse_a=*/false, /*reuse_b=*/false);
            }
            store_C(dst, c, mtile, ntile, lane, bias);
        }
        __syncthreads();

        // Attention logits over (i,k) pairs; edge@We folded in on the fly (K=5).
        for (int p = tid; p < GNN_N * GNN_K; p += NTHREADS) {
            const int i = p / GNN_K;
            const int j = s_nbr[p];
            const float* ed = &s_edge[p * 5];
            const float e0 = ed[0], e1 = ed[1], e2 = ed[2], e3 = ed[3], e4 = ed[4];
            const float* xlj = &s_xl[j * XS32];
            const float* xri = &s_xr[i * XS32];
            float acc = 0.0f;
#pragma unroll
            for (int h = 0; h < GNN_H; ++h) {
                float ew = e0 * s_We[0 * GNN_H + h] + e1 * s_We[1 * GNN_H + h]
                         + e2 * s_We[2 * GNN_H + h] + e3 * s_We[3 * GNN_H + h]
                         + e4 * s_We[4 * GNN_H + h];
                float e = xlj[h] + xri[h] + ew;
                e = (e > 0.0f) ? e : 0.2f * e;     // leaky_relu(0.2)
                acc = __fmaf_rn(e, s_att[h], acc);
            }
            s_alpha[p] = acc;
        }
        __syncthreads();

        // Softmax over K neighbors per target node.
        if (tid < GNN_N) {
            float m = -3.0e38f;
#pragma unroll
            for (int k = 0; k < GNN_K; ++k) m = fmaxf(m, s_alpha[tid * GNN_K + k]);
            float ex[GNN_K], s = 0.0f;
#pragma unroll
            for (int k = 0; k < GNN_K; ++k) {
                ex[k] = __expf(s_alpha[tid * GNN_K + k] - m);
                s += ex[k];
            }
            const float inv = 1.0f / s;
#pragma unroll
            for (int k = 0; k < GNN_K; ++k) s_alpha[tid * GNN_K + k] = ex[k] * inv;
        }
        __syncthreads();

        // Aggregate: x[i,h] = sum_k alpha * xl[nbr] + bo; relu except last layer.
        for (int idx = tid; idx < GNN_N * GNN_H; idx += NTHREADS) {
            const int i = idx >> 6, h = idx & 63;
            float v = s_bo[h];
#pragma unroll
            for (int k = 0; k < GNN_K; ++k) {
                const int j = s_nbr[i * GNN_K + k];
                v = __fmaf_rn(s_alpha[i * GNN_K + k], s_xl[j * XS32 + h], v);
            }
            if (l < 2) {
                v = fmaxf(v, 0.0f);
                s_x16[i * XS16 + h] = (_Float16)v;   // next layer's WMMA input
            } else {
                out[(size_t)(b * GNN_N + i) * GNN_H + h] = v;
            }
        }
        __syncthreads();
    }
}

// ---------------------------------------------------------------------------

extern "C" void kernel_launch(void* const* d_in, const int* in_sizes, int n_in,
                              void* d_out, int out_size, void* d_ws, size_t ws_size,
                              hipStream_t stream) {
    (void)in_sizes; (void)n_in; (void)d_ws; (void)ws_size; (void)out_size;
    const float* wdir   = (const float*)d_in[0];
    const float* wspd   = (const float*)d_in[1];
    const float* yawp   = (const float*)d_in[2];
    const float* layout = (const float*)d_in[3];
    // params: list of 3 dicts, pytree-flattened with sorted keys:
    // We, Wl, Wr, att, bl, bo, br  (7 leaves per layer)
    GParams P;
    for (int l = 0; l < 3; ++l) {
        const int base = 4 + 7 * l;
        P.We[l]  = (const float*)d_in[base + 0];
        P.Wl[l]  = (const float*)d_in[base + 1];
        P.Wr[l]  = (const float*)d_in[base + 2];
        P.att[l] = (const float*)d_in[base + 3];
        P.bl[l]  = (const float*)d_in[base + 4];
        P.bo[l]  = (const float*)d_in[base + 5];
        P.br[l]  = (const float*)d_in[base + 6];
    }
    windfarm_gnn_fused<<<GNN_B, NTHREADS, 0, stream>>>(
        wdir, wspd, yawp, layout, P, (float*)d_out);
}
